// BoxHead_82282983457444
// MI455X (gfx1250) — matrix operand order repaired and checked by reading, not compiled, not run on hardware
//
#include <hip/hip_runtime.h>
#include <hip/hip_bf16.h>

typedef __attribute__((ext_vector_type(16))) _Float16 v16h;
typedef __attribute__((ext_vector_type(8)))  _Float16 v8h;
typedef __attribute__((ext_vector_type(8)))  float    v8f;

#define MROWS 1024           // 1000 rois padded to 1024 (multiple of 128)
#define DFEAT (256*49)       // 12544
#define HID   1024
#define BPITCH 40            // LDS row pitch in halves (bank-conflict-free, 16B aligned)

// ---------------- fp32 -> f16 tiled transpose: dst[n][k] = (f16)src[k][n] ----
__global__ void transpose_f32_to_f16(const float* __restrict__ src,
                                     _Float16* __restrict__ dst,
                                     int K, int N) {
    __shared__ float tile[32][33];
    int kb = blockIdx.y * 32, nb = blockIdx.x * 32;
    int tx = threadIdx.x, ty = threadIdx.y;          // block (32, 8)
    #pragma unroll
    for (int i = 0; i < 32; i += 8) {
        int k = kb + ty + i, n = nb + tx;
        tile[ty + i][tx] = (k < K && n < N) ? src[(size_t)k * N + n] : 0.0f;
    }
    __syncthreads();
    #pragma unroll
    for (int i = 0; i < 32; i += 8) {
        int n = nb + ty + i, k = kb + tx;
        if (n < N && k < K) dst[(size_t)n * K + k] = (_Float16)tile[tx][ty + i];
    }
}

// -------- fuse Wc (1024x4) and Wr (1024x12) into Wh[16][1024] (N-major, f16) --
__global__ void build_whead(const float* __restrict__ Wc,
                            const float* __restrict__ Wr,
                            _Float16* __restrict__ Wh) {
    int idx = blockIdx.x * 256 + threadIdx.x;        // 16*1024 elements
    if (idx >= 16 * 1024) return;
    int n = idx >> 10, k = idx & 1023;
    float v = (n < 4) ? Wc[k * 4 + n] : Wr[k * 12 + (n - 4)];
    Wh[(size_t)n * 1024 + k] = (_Float16)v;
}

// ---------------- multiscale ROI align -> feat[1024][12544] f16 --------------
__global__ void roi_align(const float* __restrict__ f0, const float* __restrict__ f1,
                          const float* __restrict__ f2, const float* __restrict__ f3,
                          const float* __restrict__ proposals,
                          _Float16* __restrict__ feat) {
    int r = blockIdx.x;                  // 0..1023
    int c = threadIdx.x;                 // channel 0..255
    if (r >= 1000) {                     // zero-pad rows 1000..1023
        #pragma unroll 7
        for (int j = 0; j < 49; ++j)
            feat[(size_t)r * DFEAT + c * 49 + j] = (_Float16)0.0f;
        return;
    }
    int b = r / 500;
    const float* roi = proposals + (size_t)r * 4;
    float x1 = roi[0], y1 = roi[1], x2 = roi[2], y2 = roi[3];
    float pw = x2 - x1, ph = y2 - y1;
    float kf = floorf(4.0f + log2f(sqrtf(pw * ph) / 224.0f));
    kf = fminf(fmaxf(kf, 2.0f), 5.0f);
    int lvl = (int)kf - 2;
    const int   Hs[4] = {200, 100, 50, 25};
    const int   Ws[4] = {272, 136, 68, 34};
    const float* fm[4] = {f0, f1, f2, f3};
    int Hl = Hs[lvl], Wl = Ws[lvl];
    const float* F = fm[lvl] + ((size_t)b * 256 + c) * (size_t)(Hl * Wl);
    float sx = (float)Wl / 1088.0f, sy = (float)Hl / 800.0f;
    float rx1 = x1 * sx, ry1 = y1 * sy;
    float roiw = fmaxf(x2 * sx - rx1, 1.0f), roih = fmaxf(y2 * sy - ry1, 1.0f);
    float bw = roiw * (1.0f / 7.0f), bh = roih * (1.0f / 7.0f);
    float Hf = (float)Hl, Wf = (float)Wl;
    for (int py = 0; py < 7; ++py) {
        for (int px = 0; px < 7; ++px) {
            float sum = 0.0f;
            #pragma unroll
            for (int s = 0; s < 4; ++s) {
                float y = ry1 + ((float)py + ((s >> 1) + 0.5f) * 0.5f) * bh;
                float x = rx1 + ((float)px + ((s & 1) + 0.5f) * 0.5f) * bw;
                bool valid = (y > -1.0f) && (y < Hf) && (x > -1.0f) && (x < Wf);
                float yc = fminf(fmaxf(y, 0.0f), Hf - 1.0f);
                float xc = fminf(fmaxf(x, 0.0f), Wf - 1.0f);
                float y0f = floorf(yc), x0f = floorf(xc);
                float ly = yc - y0f, lx = xc - x0f;
                int y0 = (int)y0f, x0 = (int)x0f;
                int y1i = min(y0 + 1, Hl - 1), x1i = min(x0 + 1, Wl - 1);
                float g00 = F[y0 * Wl + x0],  g01 = F[y0 * Wl + x1i];
                float g10 = F[y1i * Wl + x0], g11 = F[y1i * Wl + x1i];
                float hy = 1.0f - ly, hx = 1.0f - lx;
                float val = hy * hx * g00 + hy * lx * g01 + ly * hx * g10 + ly * lx * g11;
                sum += valid ? val : 0.0f;
            }
            feat[(size_t)r * DFEAT + c * 49 + py * 7 + px] = (_Float16)(sum * 0.25f);
        }
    }
}

// ----- WMMA A-fragment loader, contiguous k = 16*h + s (wave32) -----------------
__device__ __forceinline__ v16h load_a_frag(const _Float16* xrow, int k0, int c8) {
    union { v16h v; v8h h[2]; } u;
    u.h[0] = *(const v8h*)(xrow + k0 + 2 * c8);        // halves 0..7  -> K = 2*c8..2*c8+7
    u.h[1] = *(const v8h*)(xrow + k0 + 2 * c8 + 8);    // halves 8..15 -> K = 2*c8+8..2*c8+15
    return u.v;
}

__device__ __forceinline__ v16h load_b_frag_lds(const _Float16* lb, int col, int k16) {
    union { v16h v; v8h h[2]; } u;
    u.h[0] = *(const v8h*)(lb + col * BPITCH + k16);
    u.h[1] = *(const v8h*)(lb + col * BPITCH + k16 + 8);
    return u.v;
}

__device__ __forceinline__ void async_copy_b128(unsigned lds_off, const _Float16* src) {
    asm volatile("global_load_async_to_lds_b128 %0, %1, off"
                 :: "v"(lds_off), "v"((unsigned long long)(uintptr_t)src)
                 : "memory");
}

// -------- GEMM: Y[M][N](f16) = relu(X[M][K](f16) @ Wt[N][K]^T + bias) --------
// block = 256 (8 waves); block tile 128 rows x 64 cols; B tile double-buffered
// through LDS via async global->LDS copies (ASYNCcnt), A direct from global.
__global__ void __launch_bounds__(256)
gemm_bias_relu(const _Float16* __restrict__ X,
               const _Float16* __restrict__ Wt,
               const float* __restrict__ bias,
               _Float16* __restrict__ Y,
               int M, int N, int K, int relu) {
    __shared__ _Float16 ldsB[2][64 * BPITCH];        // 2 x 5 KB
    int tid  = threadIdx.x;
    int lane = tid & 31;
    int wave = tid >> 5;
    int rowbase = blockIdx.y * 128 + wave * 16;
    int colbase = blockIdx.x * 64;
    int m  = lane & 15;          // A row / B col / C col within tile
    int hi = lane >> 4;          // lane-half select
    int c8  = hi << 3;           // A K-offset: 2*c8 = 0 or 16
    int k16 = hi << 4;           // B K-offset: 0 or 16

    // async-copy mapping: thread -> (col 0..63, 8-half segment 0..3)
    int ccol = tid >> 2, cseg = tid & 3;
    const _Float16* bsrc = Wt + (size_t)(colbase + ccol) * K + cseg * 8;
    unsigned lds_off0 = (unsigned)(uintptr_t)&ldsB[0][ccol * BPITCH + cseg * 8];
    unsigned lds_off1 = (unsigned)(uintptr_t)&ldsB[1][ccol * BPITCH + cseg * 8];

    int nsteps = K >> 5;
    async_copy_b128(lds_off0, bsrc);                 // prefetch step 0

    const _Float16* xrow = X + (size_t)(rowbase + m) * K;
    v8f acc0 = {}, acc1 = {}, acc2 = {}, acc3 = {};
    for (int k = 0; k < nsteps; ++k) {
        int k0 = k << 5;
        if (k + 1 < nsteps) {                        // overlap copy(k+1) with compute(k)
            async_copy_b128(((k + 1) & 1) ? lds_off1 : lds_off0, bsrc + k0 + 32);
            asm volatile("s_wait_asynccnt 0x1" ::: "memory");   // step k done (in-order)
        } else {
            asm volatile("s_wait_asynccnt 0x0" ::: "memory");
        }
        __syncthreads();                             // copies visible to all waves
        const _Float16* lb = ldsB[k & 1];
        v16h a  = load_a_frag(xrow, k0, c8);
        v16h b0 = load_b_frag_lds(lb,  0 + m, k16);
        v16h b1 = load_b_frag_lds(lb, 16 + m, k16);
        v16h b2 = load_b_frag_lds(lb, 32 + m, k16);
        v16h b3 = load_b_frag_lds(lb, 48 + m, k16);
        acc0 = __builtin_amdgcn_wmma_f32_16x16x32_f16(false, a, false, b0, (short)0, acc0, false, false);
        acc1 = __builtin_amdgcn_wmma_f32_16x16x32_f16(false, a, false, b1, (short)0, acc1, false, false);
        acc2 = __builtin_amdgcn_wmma_f32_16x16x32_f16(false, a, false, b2, (short)0, acc2, false, false);
        acc3 = __builtin_amdgcn_wmma_f32_16x16x32_f16(false, a, false, b3, (short)0, acc3, false, false);
        __syncthreads();                             // buffer reusable for copy(k+2)
    }

    v8f accs[4] = {acc0, acc1, acc2, acc3};
    #pragma unroll
    for (int t = 0; t < 4; ++t) {
        int col = colbase + t * 16 + m;
        float bv = bias[col];
        #pragma unroll
        for (int v = 0; v < 8; ++v) {
            int row = rowbase + v + hi * 8;          // C layout: M = v + 8*(lane>=16)
            float val = accs[t][v] + bv;
            if (relu && val < 0.0f) val = 0.0f;
            Y[(size_t)row * N + col] = (_Float16)val;
        }
    }
}

// -------- heads: out = h2 @ [Wc|Wr] + [bc|br]; N=16 = one WMMA tile ----------
__global__ void head_gemm(const _Float16* __restrict__ h2,
                          const _Float16* __restrict__ Wh,   // [16][1024]
                          const float* __restrict__ bc,
                          const float* __restrict__ br,
                          float* __restrict__ out, int K) {
    int lane = threadIdx.x & 31;
    int wave = threadIdx.x >> 5;
    int rowbase = blockIdx.x * 128 + wave * 16;
    int m = lane & 15, hi = lane >> 4;
    int c8 = hi << 3, k16 = hi << 4;
    const _Float16* xrow = h2 + (size_t)(rowbase + m) * K;
    v8f acc = {};
    for (int k0 = 0; k0 < K; k0 += 32) {
        v16h a = load_a_frag(xrow, k0, c8);
        union { v16h v; v8h h[2]; } b;
        b.h[0] = *(const v8h*)(Wh + (size_t)m * K + k0 + k16);
        b.h[1] = *(const v8h*)(Wh + (size_t)m * K + k0 + k16 + 8);
        acc = __builtin_amdgcn_wmma_f32_16x16x32_f16(false, a, false, b.v, (short)0, acc, false, false);
    }
    int n = m;                                   // output column 0..15
    float bias = (n < 4) ? bc[n] : br[n - 4];
    #pragma unroll
    for (int v = 0; v < 8; ++v) {
        int row = rowbase + v + hi * 8;
        if (row < 1000) {
            float val = acc[v] + bias;
            if (n < 4) out[row * 4 + n] = val;                       // class_logits
            else       out[4000 + row * 12 + (n - 4)] = val;         // box_regression
        }
    }
}

extern "C" void kernel_launch(void* const* d_in, const int* in_sizes, int n_in,
                              void* d_out, int out_size, void* d_ws, size_t ws_size,
                              hipStream_t stream) {
    const float* fpn0 = (const float*)d_in[0];
    const float* fpn1 = (const float*)d_in[1];
    const float* fpn2 = (const float*)d_in[2];
    const float* fpn3 = (const float*)d_in[3];
    const float* prop = (const float*)d_in[4];
    const float* W1   = (const float*)d_in[5];
    const float* b1   = (const float*)d_in[6];
    const float* W2   = (const float*)d_in[7];
    const float* b2   = (const float*)d_in[8];
    const float* Wc   = (const float*)d_in[9];
    const float* bc   = (const float*)d_in[10];
    const float* Wr   = (const float*)d_in[11];
    const float* br   = (const float*)d_in[12];
    float* out = (float*)d_out;

    _Float16* ws = (_Float16*)d_ws;
    size_t oWt1  = 0;                                   // [1024][12544]
    size_t oFeat = oWt1  + (size_t)HID * DFEAT;         // [1024][12544]
    size_t oWt2  = oFeat + (size_t)MROWS * DFEAT;       // [1024][1024]
    size_t oH1   = oWt2  + (size_t)HID * HID;           // [1024][1024]
    size_t oH2   = oH1   + (size_t)MROWS * HID;         // [1024][1024]
    size_t oWh   = oH2   + (size_t)MROWS * HID;         // [16][1024]
    _Float16 *Wt1 = ws + oWt1, *feat = ws + oFeat, *Wt2 = ws + oWt2;
    _Float16 *h1 = ws + oH1, *h2 = ws + oH2, *Wh = ws + oWh;

    // weight conversion: fp32 row-major -> f16 N-major (L2-resident afterwards)
    transpose_f32_to_f16<<<dim3(HID / 32, DFEAT / 32), dim3(32, 8), 0, stream>>>(W1, Wt1, DFEAT, HID);
    transpose_f32_to_f16<<<dim3(HID / 32, HID / 32),   dim3(32, 8), 0, stream>>>(W2, Wt2, HID, HID);
    build_whead<<<64, 256, 0, stream>>>(Wc, Wr, Wh);

    // ROI align -> padded f16 feature matrix
    roi_align<<<MROWS, 256, 0, stream>>>(fpn0, fpn1, fpn2, fpn3, prop, feat);

    // MLP via WMMA (f32 accumulate), async B staging, relu fused
    gemm_bias_relu<<<dim3(HID / 64, MROWS / 128), 256, 0, stream>>>(feat, Wt1, b1, h1, MROWS, HID, DFEAT, 1);
    gemm_bias_relu<<<dim3(HID / 64, MROWS / 128), 256, 0, stream>>>(h1,   Wt2, b2, h2, MROWS, HID, HID,   1);

    // heads (4 + 12 = one 16-wide WMMA tile), scatter to d_out
    head_gemm<<<MROWS / 128, 256, 0, stream>>>(h2, Wh, bc, br, out, HID);
}
